// WassersteinLoss_6485400617246
// MI455X (gfx1250) — compile-verified, matrix operand order checked by reading, not run on hardware
//
#include <hip/hip_runtime.h>

typedef __attribute__((ext_vector_type(16))) _Float16 v16h;
typedef __attribute__((ext_vector_type(8)))  float    v8f;

#define B_N 4096
#define D_K 1024
#define EPS_W 0.05f
#define REG_W 0.1f
#define NUM_ITER_W 10
#define INV_B (1.0f / 4096.0f)

__device__ __forceinline__ float waveReduceSum(float x) {
    for (int off = 16; off > 0; off >>= 1) x += __shfl_down(x, off, 32);
    return x;
}
__device__ __forceinline__ float waveReduceMax(float x) {
    for (int off = 16; off > 0; off >>= 1) x = fmaxf(x, __shfl_down(x, off, 32));
    return x;
}

// ---------------- init scratch state (fresh every call) ----------------
__global__ void k_init(float* u, float* v, float* scal) {
    int i = blockIdx.x * blockDim.x + threadIdx.x;
    if (i < B_N) { u[i] = INV_B; v[i] = INV_B; }
    if (i == 0) {
        ((unsigned*)scal)[0] = 0u;  // max(M) bits
        scal[1] = 0.f;              // ce accumulator
        scal[2] = 0.f;              // kl_col accumulator
        scal[3] = 0.f;              // kl_row accumulator
    }
}

// ---------------- row-normalize both embeddings to f16 ----------------
__global__ __launch_bounds__(256) void k_normalize(const float* __restrict__ audio,
                                                   const float* __restrict__ text,
                                                   _Float16* __restrict__ An,
                                                   _Float16* __restrict__ Tn) {
    int wave = threadIdx.x >> 5, lane = threadIdx.x & 31;
    int row = blockIdx.x * 8 + wave;                       // 0..8191
    const float* src = (row < B_N) ? (audio + (size_t)row * D_K)
                                   : (text + (size_t)(row - B_N) * D_K);
    _Float16* dst = (row < B_N) ? (An + (size_t)row * D_K)
                                : (Tn + (size_t)(row - B_N) * D_K);
    float ss = 0.f;
    for (int j = lane; j < D_K; j += 32) { float x = src[j]; ss += x * x; }
    ss = waveReduceSum(ss);
    ss = __shfl(ss, 0, 32);
    float inv = rsqrtf(ss);
    for (int j = lane; j < D_K; j += 32) dst[j] = (_Float16)(src[j] * inv);
}

// ---------------- WMMA GEMM: M = 1 - An @ Tn^T, track max(M) ----------------
// Double-buffered LDS; tiles staged with CDNA5 async global->LDS copies
// (GLOBAL_LOAD_ASYNC_TO_LDS_B128, ASYNCcnt) so the copy of chunk k+1
// overlaps the WMMAs of chunk k without burning staging VGPRs.
__global__ __launch_bounds__(256) void k_gemm(const _Float16* __restrict__ An,
                                              const _Float16* __restrict__ Tn,
                                              float* __restrict__ Mb,
                                              unsigned* __restrict__ maxBits) {
    __shared__ _Float16 As[2][128][40];   // 128 rows x 32 k-halves (+8 pad), x2 buffers
    __shared__ _Float16 Bs[2][128][40];
    __shared__ float wmax[8];

    const int i0 = blockIdx.x * 128, j0 = blockIdx.y * 128;
    const int tid  = threadIdx.x;
    const int wave = tid >> 5, lane = tid & 31;
    const int lidx = lane & 15, lhi = lane >> 4;
    const int wr = (wave >> 1) * 32;   // wave's row offset in tile
    const int wc = (wave & 1) * 64;    // wave's col offset in tile

    const _Float16* Abase = An + (size_t)i0 * D_K;
    const _Float16* Bbase = Tn + (size_t)j0 * D_K;

    const v8f vzero = {0.f, 0.f, 0.f, 0.f, 0.f, 0.f, 0.f, 0.f};
    v8f acc[2][4];
    for (int a = 0; a < 2; ++a)
        for (int b = 0; b < 4; ++b) acc[a][b] = vzero;

    union Frag { v16h v; uint4 q[2]; };

    // async-stage one 128x32-half tile of A and B into LDS buffer `buf`
    auto issueTiles = [&](int buf, int kc) {
        for (int it = 0; it < 2; ++it) {
            int lin  = it * 256 + tid;
            int rrow = lin >> 2, seg = lin & 3;
            unsigned aoff = (unsigned)(uintptr_t)&As[buf][rrow][seg * 8];
            unsigned long long ga =
                (unsigned long long)(uintptr_t)(Abase + (size_t)rrow * D_K + kc + seg * 8);
            asm volatile("global_load_async_to_lds_b128 %0, %1, off"
                         :: "v"(aoff), "v"(ga) : "memory");
            unsigned boff = (unsigned)(uintptr_t)&Bs[buf][rrow][seg * 8];
            unsigned long long gb =
                (unsigned long long)(uintptr_t)(Bbase + (size_t)rrow * D_K + kc + seg * 8);
            asm volatile("global_load_async_to_lds_b128 %0, %1, off"
                         :: "v"(boff), "v"(gb) : "memory");
        }
    };

    issueTiles(0, 0);
    for (int kc = 0; kc < D_K; kc += 32) {
        int cur = (kc >> 5) & 1;
        // drain this wave's async writes, then barrier => whole tile visible
        asm volatile("s_wait_asynccnt 0" ::: "memory");
        __syncthreads();
        if (kc + 32 < D_K) issueTiles(cur ^ 1, kc + 32);   // prefetch next chunk

        // assemble fragments: lanes 0-15 take K{0..7,16..23}, lanes 16-31 take K{8..15,24..31}
        Frag a[2], b[4];
        for (int rr = 0; rr < 2; ++rr) {
            const _Float16* p = &As[cur][wr + rr * 16 + lidx][lhi * 8];
            a[rr].q[0] = *(const uint4*)p;
            a[rr].q[1] = *(const uint4*)(p + 16);
        }
        for (int qq = 0; qq < 4; ++qq) {
            const _Float16* p = &Bs[cur][wc + qq * 16 + lidx][lhi * 8];
            b[qq].q[0] = *(const uint4*)p;
            b[qq].q[1] = *(const uint4*)(p + 16);
        }
        for (int rr = 0; rr < 2; ++rr)
            for (int qq = 0; qq < 4; ++qq)
                acc[rr][qq] = __builtin_amdgcn_wmma_f32_16x16x32_f16(
                    false, a[rr].v, false, b[qq].v, (short)0, acc[rr][qq], false, false);
    }

    // epilogue: M = 1 - c ; C/D layout: VGPR p -> M = p + lhi*8, N = lidx
    float localMax = 0.f;
    for (int rr = 0; rr < 2; ++rr)
        for (int qq = 0; qq < 4; ++qq)
            for (int p = 0; p < 8; ++p) {
                float m = 1.0f - acc[rr][qq][p];
                int gi = i0 + wr + rr * 16 + p + lhi * 8;
                int gj = j0 + wc + qq * 16 + lidx;
                Mb[(size_t)gi * B_N + gj] = m;
                localMax = fmaxf(localMax, m);
            }
    localMax = fmaxf(localMax, 0.f);
    localMax = waveReduceMax(localMax);
    if (lane == 0) wmax[wave] = localMax;
    __syncthreads();
    if (tid == 0) {
        float bm = wmax[0];
        for (int w = 1; w < 8; ++w) bm = fmaxf(bm, wmax[w]);
        atomicMax(maxBits, __float_as_uint(bm));   // monotonic for non-negative floats
    }
}

// ---------------- K = exp(-M / (maxM * eps)), in place ----------------
__global__ void k_expk(float* __restrict__ Km, const unsigned* __restrict__ maxBits, size_t n) {
    float mmax = __uint_as_float(maxBits[0]);
    float s = -1.0f / (mmax * EPS_W);
    size_t stride = (size_t)gridDim.x * blockDim.x;
    for (size_t i = blockIdx.x * (size_t)blockDim.x + threadIdx.x; i < n; i += stride)
        Km[i] = __expf(Km[i] * s);
}

// ---------------- t[j] = sum_i K[i][j]*u[i]  (row-major, column-block per WG) ----------------
__global__ __launch_bounds__(256) void k_colsum(const float* __restrict__ Km,
                                                const float* __restrict__ u,
                                                float* __restrict__ out, int divide) {
    __shared__ float red[256];
    int j0 = blockIdx.x * 64;
    int jl = threadIdx.x & 63, ig = threadIdx.x >> 6;
    float part = 0.f;
    for (int i = ig; i < B_N; i += 4)
        part += Km[(size_t)i * B_N + j0 + jl] * u[i];
    red[threadIdx.x] = part;
    __syncthreads();
    if (threadIdx.x < 64) {
        float s = red[threadIdx.x] + red[threadIdx.x + 64] +
                  red[threadIdx.x + 128] + red[threadIdx.x + 192];
        out[j0 + threadIdx.x] = divide ? (INV_B / s) : s;
    }
}

// ---------------- r[i] = sum_j K[i][j]*v[j]  (wave per row) ----------------
__global__ __launch_bounds__(256) void k_rowdot(const float* __restrict__ Km,
                                                const float* __restrict__ vin,
                                                float* __restrict__ out, int divide) {
    __shared__ float sv[B_N];
    for (int j = threadIdx.x; j < B_N; j += 256) sv[j] = vin[j];
    __syncthreads();
    int wave = threadIdx.x >> 5, lane = threadIdx.x & 31;
    for (int p = 0; p < 8; ++p) {
        int row = blockIdx.x * 64 + p * 8 + wave;
        const float* kr = Km + (size_t)row * B_N;
        float part = 0.f;
        for (int j = lane; j < B_N; j += 32) part += kr[j] * sv[j];
        part = waveReduceSum(part);
        if (lane == 0) out[row] = divide ? (INV_B / part) : part;
    }
}

// ---------------- cross entropy over rows of pi = u * K * v^T ----------------
__global__ __launch_bounds__(256) void k_ce(const float* __restrict__ Km,
                                            const float* __restrict__ u,
                                            const float* __restrict__ v,
                                            float* __restrict__ scal) {
    __shared__ float sv[B_N];
    for (int j = threadIdx.x; j < B_N; j += 256) sv[j] = v[j];
    __syncthreads();
    int wave = threadIdx.x >> 5, lane = threadIdx.x & 31;
    float waveAcc = 0.f;
    for (int p = 0; p < 8; ++p) {
        int row = blockIdx.x * 64 + p * 8 + wave;
        const float* kr = Km + (size_t)row * B_N;
        float ui = u[row];
        float m = -3.4e38f;
        for (int j = lane; j < B_N; j += 32) m = fmaxf(m, ui * kr[j] * sv[j]);
        m = waveReduceMax(m);
        m = __shfl(m, 0, 32);
        float s = 0.f;
        for (int j = lane; j < B_N; j += 32) s += __expf(ui * kr[j] * sv[j] - m);
        s = waveReduceSum(s);
        if (lane == 0) {
            float diag = ui * kr[row] * sv[row];
            waveAcc += -(diag - m - logf(s));   // -logp[row][row]
        }
    }
    if (lane == 0) atomicAdd(&scal[1], waveAcc);
}

// ---------------- KL terms on col = v*t and row = u*r ----------------
__global__ void k_kl(const float* __restrict__ v, const float* __restrict__ t,
                     const float* __restrict__ u, const float* __restrict__ r,
                     float* __restrict__ scal) {
    int i = blockIdx.x * blockDim.x + threadIdx.x;
    if (i >= B_N) return;
    float cj = v[i] * t[i];
    float kc = (cj > 0.f) ? cj * (logf(cj) - INV_B) : 0.f;
    float ri = u[i] * r[i];
    float kr = (ri > 0.f) ? ri * (logf(ri) - INV_B) : 0.f;
    atomicAdd(&scal[2], kc);
    atomicAdd(&scal[3], kr);
}

__global__ void k_finalize(const float* __restrict__ scal, float* __restrict__ out) {
    if (threadIdx.x == 0 && blockIdx.x == 0)
        out[0] = scal[1] * INV_B + REG_W * (scal[2] * INV_B + scal[3] * INV_B);
}

extern "C" void kernel_launch(void* const* d_in, const int* in_sizes, int n_in,
                              void* d_out, int out_size, void* d_ws, size_t ws_size,
                              hipStream_t stream) {
    const float* audio = (const float*)d_in[0];
    const float* text  = (const float*)d_in[1];
    (void)in_sizes; (void)n_in; (void)out_size; (void)ws_size;

    char* ws = (char*)d_ws;
    size_t off = 0;
    _Float16* An = (_Float16*)(ws + off); off += (size_t)B_N * D_K * sizeof(_Float16); // 8 MB
    _Float16* Tn = (_Float16*)(ws + off); off += (size_t)B_N * D_K * sizeof(_Float16); // 8 MB
    float* Km = (float*)(ws + off);       off += (size_t)B_N * B_N * sizeof(float);    // 64 MB (M -> K in place)
    float* u  = (float*)(ws + off);       off += (size_t)B_N * sizeof(float);
    float* v  = (float*)(ws + off);       off += (size_t)B_N * sizeof(float);
    float* t  = (float*)(ws + off);       off += (size_t)B_N * sizeof(float);
    float* r  = (float*)(ws + off);       off += (size_t)B_N * sizeof(float);
    float* scal = (float*)(ws + off);     // [0]=maxM bits, [1]=ce, [2]=kl_col, [3]=kl_row

    k_init<<<16, 256, 0, stream>>>(u, v, scal);
    k_normalize<<<1024, 256, 0, stream>>>(audio, text, An, Tn);
    k_gemm<<<dim3(32, 32), 256, 0, stream>>>(An, Tn, Km, (unsigned*)scal);
    k_expk<<<4096, 256, 0, stream>>>(Km, (const unsigned*)scal, (size_t)B_N * B_N);

    for (int it = 0; it < NUM_ITER_W; ++it) {
        k_colsum<<<64, 256, 0, stream>>>(Km, u, v, 1);   // v = b / (K^T u)
        k_rowdot<<<64, 256, 0, stream>>>(Km, v, u, 1);   // u = a / (K v)
    }
    k_colsum<<<64, 256, 0, stream>>>(Km, u, t, 0);       // t = K^T u (final)
    k_rowdot<<<64, 256, 0, stream>>>(Km, v, r, 0);       // r = K v   (final)
    k_ce<<<64, 256, 0, stream>>>(Km, u, v, scal);
    k_kl<<<16, 256, 0, stream>>>(v, t, u, r, scal);
    k_finalize<<<1, 64, 0, stream>>>(scal, (float*)d_out);
}
